// CustomS4_79027398246823
// MI455X (gfx1250) — matrix operand order repaired and checked by reading, not compiled, not run on hardware
//
#include <hip/hip_runtime.h>
#include <hip/hip_bf16.h>

// CDNA5 (gfx1250) fused S4 pipeline:
//  - bf16 WMMA (v_wmma_f32_16x16x32_bf16) fused GEMM: y=x@W^T (stats only) | v=x@H
//  - LayerNorm folded algebraically into the N=64 projection
//  - linear scan decomposed into parallel chunk GEMMs + log-depth power chain
//  - double-buffered LDS x-staging, batched B-fragment loads for latency hiding

typedef __attribute__((ext_vector_type(16))) __bf16        v16bf;
typedef __attribute__((ext_vector_type(8)))  float         v8f;
typedef __attribute__((ext_vector_type(8)))  unsigned int  v8u;

__device__ __forceinline__ unsigned short f2bf(float f) {
    unsigned int u = __float_as_uint(f);
    unsigned int r = u + 0x7FFFu + ((u >> 16) & 1u);   // round-to-nearest-even
    return (unsigned short)(r >> 16);
}

__device__ __forceinline__ v16bf mkfrag(uint4 lo, uint4 hi) {
    v8u t;
    t[0] = lo.x; t[1] = lo.y; t[2] = lo.z; t[3] = lo.w;
    t[4] = hi.x; t[5] = hi.y; t[6] = hi.z; t[7] = hi.w;
    return __builtin_bit_cast(v16bf, t);
}

// ---------------------------------------------------------------- prep: gv / cv / bG
__global__ __launch_bounds__(192) void prep_vec(
    const float* __restrict__ bvec, const float* __restrict__ gamma,
    const float* __restrict__ beta, const float* __restrict__ Bm,
    float* __restrict__ vecs)
{
    int t = threadIdx.x;
    int n = t & 63, which = t >> 6;
    float s = 0.f;
    for (int d = 0; d < 768; ++d) {
        float w = (which == 0) ? gamma[d] : (which == 1 ? beta[d] : bvec[d] * gamma[d]);
        s += w * Bm[d * 64 + n];
    }
    vecs[which * 64 + n] = s;
}

// ---------------------------------------------------------------- prep: H = W^T diag(gamma) Bm  (4-way K-split partials)
__global__ __launch_bounds__(256) void prep_H(
    const float* __restrict__ W, const float* __restrict__ gamma,
    const float* __restrict__ Bm, float* __restrict__ Hp)
{
    __shared__ float wl[64 * 65];
    __shared__ float gb[64 * 65];
    int tid = threadIdx.x;
    int es = blockIdx.x / 12, dt = blockIdx.x % 12;
    int d = tid & 63, ng = tid >> 6;
    float acc[16];
#pragma unroll
    for (int i = 0; i < 16; ++i) acc[i] = 0.f;
    for (int cc = 0; cc < 3; ++cc) {
        int e0 = (es * 3 + cc) * 64;
        __syncthreads();
        for (int it = 0; it < 16; ++it) {
            int idx = it * 256 + tid;
            int el = idx >> 6, q = idx & 63;
            wl[el * 65 + q] = W[(e0 + el) * 768 + dt * 64 + q];
            gb[el * 65 + q] = gamma[e0 + el] * Bm[(e0 + el) * 64 + q];
        }
        __syncthreads();
        for (int el = 0; el < 64; ++el) {
            float wv = wl[el * 65 + d];
#pragma unroll
            for (int i = 0; i < 16; ++i) acc[i] += wv * gb[el * 65 + ng * 16 + i];
        }
    }
    for (int i = 0; i < 16; ++i)
        Hp[es * 49152 + (dt * 64 + d) * 64 + ng * 16 + i] = acc[i];
}

// ---------------------------------------------------------------- prep: pack B = [W^T | H] into per-lane WMMA fragments (bf16)
// B layout used here (contiguous K, 16-bit, 32x16): lane = n + 16*(k/16); dword = (k%16)/2; half = k%2
__global__ __launch_bounds__(256) void prep_bfrag(
    const float* __restrict__ W, const float* __restrict__ Hp,
    unsigned int* __restrict__ bfrag)
{
    int idx = blockIdx.x * 256 + threadIdx.x;          // < 24*52*32*8 = 319488
    int dword = idx & 7;
    int lane  = (idx >> 3) & 31;
    int nt    = (idx >> 8) % 52;
    int kt    = idx / 13312;
    int part = lane >> 4, nn = lane & 15;
    int n_g = nt * 16 + nn;
    unsigned int outv = 0;
#pragma unroll
    for (int h = 0; h < 2; ++h) {
        int k_g = kt * 32 + part * 16 + dword * 2 + h;
        float v;
        if (n_g < 768) {
            v = W[n_g * 768 + k_g];                    // W^T[k][e] = W[e][k]
        } else {
            int n2 = n_g - 768;
            v = Hp[k_g * 64 + n2] + Hp[49152 + k_g * 64 + n2]
              + Hp[2 * 49152 + k_g * 64 + n2] + Hp[3 * 49152 + k_g * 64 + n2];
        }
        outv |= (unsigned int)f2bf(v) << (16 * h);
    }
    bfrag[idx] = outv;
}

// ---------------------------------------------------------------- power chain: Ppow[i] = A^i  (log-depth doubling)
__global__ __launch_bounds__(256) void mat_init(const float* __restrict__ A, float* __restrict__ Ppow)
{
    int t = blockIdx.x * 256 + threadIdx.x;            // 8192 elements
    if (t < 4096) Ppow[t] = ((t >> 6) == (t & 63)) ? 1.f : 0.f;   // A^0 = I
    else          Ppow[t] = A[t - 4096];                          // A^1
}

__global__ __launch_bounds__(256) void mat_mul_batch(
    const float* __restrict__ srcA, const float* __restrict__ srcB, float* __restrict__ dst)
{
    __shared__ float al[64 * 66];
    __shared__ float bl[64 * 65];
    int tid = threadIdx.x;
    const float* Am = srcA + (long)blockIdx.x * 4096;
    float*       D  = dst  + (long)blockIdx.x * 4096;
    for (int it = 0; it < 16; ++it) {
        int idx = it * 256 + tid;
        int r = idx >> 6, c = idx & 63;
        al[r * 66 + c] = Am[idx];
        bl[r * 65 + c] = srcB[idx];
    }
    __syncthreads();
    int c = tid & 63, rb = (tid >> 6) << 4;
    float acc[16];
#pragma unroll
    for (int i = 0; i < 16; ++i) acc[i] = 0.f;
    for (int k = 0; k < 64; ++k) {
        float bv = bl[k * 65 + c];
#pragma unroll
        for (int i = 0; i < 16; ++i) acc[i] += al[(rb + i) * 66 + k] * bv;
    }
    for (int i = 0; i < 16; ++i) D[(rb + i) * 64 + c] = acc[i];
}

// ---------------------------------------------------------------- fused main GEMM (WMMA bf16) + LN-stats + u projection
__global__ __launch_bounds__(512) void fused_gemm(
    const float* __restrict__ x, const float* __restrict__ bvec,
    const unsigned int* __restrict__ bfrag,
    const float* __restrict__ gv, const float* __restrict__ cv, const float* __restrict__ bG,
    unsigned short* __restrict__ u_out)
{
    __shared__ unsigned short xs[2][64 * 48];          // double-buffered 64x32 bf16 slab, stride 48
    __shared__ float s_sum[64], s_sq[64], s_mean[64], s_rstd[64];
    int tid = threadIdx.x;
    int w = tid >> 5, lane = tid & 31;
    int col = lane & 15, halfId = lane >> 4;
    long block0 = (long)blockIdx.x * 64;

    v8f acc[4][4];
#pragma unroll
    for (int k = 0; k < 4; ++k)
#pragma unroll
        for (int mt = 0; mt < 4; ++mt) { v8f z = {}; acc[k][mt] = z; }

    if (tid < 64) { s_sum[tid] = 0.f; s_sq[tid] = 0.f; }

    int srow = tid >> 3, sc4 = (tid & 7) << 2;
    const float* xrow = x + (block0 + srow) * 768 + sc4;

    {   // prologue: stage slab 0
        const float4 f = *(const float4*)(xrow);
        unsigned int d0 = (unsigned int)f2bf(f.x) | ((unsigned int)f2bf(f.y) << 16);
        unsigned int d1 = (unsigned int)f2bf(f.z) | ((unsigned int)f2bf(f.w) << 16);
        *(uint2*)&xs[0][srow * 48 + sc4] = make_uint2(d0, d1);
    }

    for (int kt = 0; kt < 24; ++kt) {
        __syncthreads();                               // xs[kt&1] ready; xs[(kt+1)&1] free

        // issue next slab's global load early (HBM stream overlaps WMMA)
        float4 fn;
        bool have_next = (kt + 1 < 24);
        if (have_next) fn = *(const float4*)(xrow + (kt + 1) * 32);
        if (kt + 2 < 24)
            __builtin_prefetch(xrow + (kt + 2) * 32, 0, 1);

        // A fragments (contiguous K, matching bfrag): lanes 0-15 K 0-15, lanes 16-31 K 16-31
        const unsigned short* xb = &xs[kt & 1][0];
        uint4 alo[4], ahi[4];
        int koff = halfId * 16;
#pragma unroll
        for (int mt = 0; mt < 4; ++mt) {
            const unsigned short* p = &xb[(mt * 16 + col) * 48 + koff];
            alo[mt] = *(const uint4*)p;
            ahi[mt] = *(const uint4*)(p + 8);
        }

        // batch-load all owned B fragments (one load clause), then WMMA block
        uint4 blo[4], bhi[4];
#pragma unroll
        for (int k = 0; k < 4; ++k) {
            if (k < 3 || w < 4) {
                const uint4* bp = (const uint4*)(bfrag + (((kt * 52 + (w + 16 * k)) * 32 + lane) << 3));
                blo[k] = bp[0];
                bhi[k] = bp[1];
            }
        }
#pragma unroll
        for (int k = 0; k < 4; ++k) {
            if (k < 3 || w < 4) {                      // waves 0-3 also own the H (v) tile
                v16bf bf = mkfrag(blo[k], bhi[k]);
#pragma unroll
                for (int mt = 0; mt < 4; ++mt) {
                    v16bf af = mkfrag(alo[mt], ahi[mt]);
                    acc[k][mt] = __builtin_amdgcn_wmma_f32_16x16x32_bf16(
                        false, af, false, bf, (short)0, acc[k][mt], false, false);
                }
            }
        }

        // convert + store next slab into the alternate buffer
        if (have_next) {
            unsigned int d0 = (unsigned int)f2bf(fn.x) | ((unsigned int)f2bf(fn.y) << 16);
            unsigned int d1 = (unsigned int)f2bf(fn.z) | ((unsigned int)f2bf(fn.w) << 16);
            *(uint2*)&xs[(kt + 1) & 1][srow * 48 + sc4] = make_uint2(d0, d1);
        }
    }

    // row stats over y tiles (k=0..2 for every wave), bias included
    float bb[3];
#pragma unroll
    for (int k = 0; k < 3; ++k) bb[k] = bvec[(w + 16 * k) * 16 + col];
#pragma unroll
    for (int mt = 0; mt < 4; ++mt) {
#pragma unroll
        for (int r = 0; r < 8; ++r) {
            float s = 0.f, q = 0.f;
#pragma unroll
            for (int k = 0; k < 3; ++k) {
                float y = acc[k][mt][r] + bb[k];
                s += y; q += y * y;
            }
#pragma unroll
            for (int m = 1; m < 16; m <<= 1) { s += __shfl_xor(s, m); q += __shfl_xor(q, m); }
            if (col == 0) {
                int row = mt * 16 + halfId * 8 + r;
                atomicAdd(&s_sum[row], s);
                atomicAdd(&s_sq[row], q);
            }
        }
    }
    __syncthreads();
    if (tid < 64) {
        float mean = s_sum[tid] * (1.f / 768.f);
        float var  = s_sq[tid] * (1.f / 768.f) - mean * mean;
        s_mean[tid] = mean;
        s_rstd[tid] = rsqrtf(var + 1e-5f);
    }
    __syncthreads();

    // u = rstd*(v + bG) - rstd*mean*gv + cv   (waves 0-3 hold the H columns)
    if (w < 4) {
        int n = w * 16 + col;
        float gvn = gv[n], cvn = cv[n], bGn = bG[n];
#pragma unroll
        for (int mt = 0; mt < 4; ++mt) {
#pragma unroll
            for (int r = 0; r < 8; ++r) {
                int row = mt * 16 + halfId * 8 + r;
                float v = acc[3][mt][r] + bGn;
                float uu = s_rstd[row] * (v - s_mean[row] * gvn) + cvn;
                u_out[(block0 + row) * 64 + n] = f2bf(uu);
            }
        }
    }
}

// ---------------------------------------------------------------- chunk partials: ps[s][j][b][n] (f32, split-K over 4 timesteps)
__global__ __launch_bounds__(256) void chunk_gemm(
    const unsigned short* __restrict__ u, const float* __restrict__ Ppow,
    float* __restrict__ ps)
{
    __shared__ float us[32 * 260];
    int tid = threadIdx.x;
    int j = blockIdx.x >> 4, s = blockIdx.x & 15;
    const unsigned int* up = (const unsigned int*)u;
    for (int it = 0; it < 16; ++it) {
        int idx = it * 256 + tid;                      // 4096 dwords
        int b = idx >> 7, dcol = idx & 127;
        int hk = dcol << 1;
        int il = hk >> 6;
        long rowg = (long)b * 2048 + j * 64 + s * 4 + il;
        unsigned int v = up[rowg * 32 + ((hk & 63) >> 1)];
        us[b * 260 + hk]     = __uint_as_float((v & 0xFFFFu) << 16);
        us[b * 260 + hk + 1] = __uint_as_float(v & 0xFFFF0000u);
    }
    __syncthreads();
    int n = tid & 63, g = tid >> 6;
    float acc[8];
#pragma unroll
    for (int i = 0; i < 8; ++i) acc[i] = 0.f;
    for (int kk = 0; kk < 256; ++kk) {
        int il = kk >> 6, kr = kk & 63;
        float p = Ppow[(63 - (s * 4 + il)) * 4096 + kr * 64 + n];
#pragma unroll
        for (int b2 = 0; b2 < 8; ++b2)
            acc[b2] += us[(g * 8 + b2) * 260 + kk] * p;
    }
#pragma unroll
    for (int b2 = 0; b2 < 8; ++b2)
        ps[(((long)s * 32 + j) * 32 + (g * 8 + b2)) * 64 + n] = acc[b2];
}

// ---------------------------------------------------------------- final: carry combine (32 steps) + h@C + L2 normalize
__global__ __launch_bounds__(1024) void final_combine(
    const float* __restrict__ ps, const float* __restrict__ A64s,
    const float* __restrict__ C, float* __restrict__ out)
{
    __shared__ float h[2][32 * 68];
    __shared__ float a64[4096];
    int tid = threadIdx.x;
    for (int i = tid; i < 4096; i += 1024) a64[i] = A64s[i];
#pragma unroll
    for (int t2 = 0; t2 < 2; ++t2) {
        int e = tid + t2 * 1024;
        h[0][(e >> 6) * 68 + (e & 63)] = 0.f;
    }
    __syncthreads();
    int cur = 0;
    for (int j = 0; j < 32; ++j) {
#pragma unroll
        for (int t2 = 0; t2 < 2; ++t2) {
            int e = tid + t2 * 1024;
            int b = e >> 6, n = e & 63;
            float accv = 0.f;
            for (int s = 0; s < 16; ++s) accv += ps[(((long)s * 32 + j) * 32 + b) * 64 + n];
            for (int k2 = 0; k2 < 64; ++k2) accv += h[cur][b * 68 + k2] * a64[k2 * 64 + n];
            h[cur ^ 1][b * 68 + n] = accv;
        }
        __syncthreads();
        cur ^= 1;
    }
    int b = tid >> 5, ln = tid & 31;
    float o[24];
#pragma unroll
    for (int i = 0; i < 24; ++i) o[i] = 0.f;
    for (int k2 = 0; k2 < 64; ++k2) {
        float hv = h[cur][b * 68 + k2];
#pragma unroll
        for (int i = 0; i < 24; ++i) o[i] += hv * C[k2 * 768 + ln + i * 32];
    }
    float sq = 0.f;
#pragma unroll
    for (int i = 0; i < 24; ++i) sq += o[i] * o[i];
#pragma unroll
    for (int m = 1; m < 32; m <<= 1) sq += __shfl_xor(sq, m);
    float inv = 1.f / fmaxf(sqrtf(sq), 1e-12f);
#pragma unroll
    for (int i = 0; i < 24; ++i) out[b * 768 + ln + i * 32] = o[i] * inv;
}

// ---------------------------------------------------------------- host
extern "C" void kernel_launch(void* const* d_in, const int* in_sizes, int n_in,
                              void* d_out, int out_size, void* d_ws, size_t ws_size,
                              hipStream_t stream) {
    (void)in_sizes; (void)n_in; (void)out_size; (void)ws_size;
    const float* x     = (const float*)d_in[0];
    const float* W     = (const float*)d_in[1];
    const float* bvec  = (const float*)d_in[2];
    const float* gamma = (const float*)d_in[3];
    const float* beta  = (const float*)d_in[4];
    const float* A     = (const float*)d_in[5];
    const float* Bm    = (const float*)d_in[6];
    const float* C     = (const float*)d_in[7];
    float* out = (float*)d_out;

    char* ws = (char*)d_ws;
    size_t off = 0;
    auto alloc = [&](size_t bytes) { void* p = ws + off; off = (off + bytes + 255) & ~(size_t)255; return p; };
    float*          Ppow  = (float*)alloc((size_t)64 * 4096 * 4);
    float*          A64   = (float*)alloc((size_t)4096 * 4);
    float*          Hp    = (float*)alloc((size_t)4 * 768 * 64 * 4);
    float*          vecs  = (float*)alloc((size_t)3 * 64 * 4);
    unsigned int*   bfrag = (unsigned int*)alloc((size_t)24 * 52 * 32 * 8 * 4);
    unsigned short* u     = (unsigned short*)alloc((size_t)65536 * 64 * 2);
    float*          ps    = (float*)alloc((size_t)16 * 32 * 32 * 64 * 4);

    prep_vec  <<<1,    192, 0, stream>>>(bvec, gamma, beta, Bm, vecs);
    prep_H    <<<48,   256, 0, stream>>>(W, gamma, Bm, Hp);
    prep_bfrag<<<1248, 256, 0, stream>>>(W, Hp, bfrag);
    mat_init  <<<32,   256, 0, stream>>>(A, Ppow);
    // doubling ladder: Ppow[L+1+k] = A^{1+k} @ A^{L}
    mat_mul_batch<<<1,  256, 0, stream>>>(Ppow + 4096, Ppow + 1  * 4096, Ppow + 2  * 4096); // P2
    mat_mul_batch<<<2,  256, 0, stream>>>(Ppow + 4096, Ppow + 2  * 4096, Ppow + 3  * 4096); // P3..P4
    mat_mul_batch<<<4,  256, 0, stream>>>(Ppow + 4096, Ppow + 4  * 4096, Ppow + 5  * 4096); // P5..P8
    mat_mul_batch<<<8,  256, 0, stream>>>(Ppow + 4096, Ppow + 8  * 4096, Ppow + 9  * 4096); // P9..P16
    mat_mul_batch<<<16, 256, 0, stream>>>(Ppow + 4096, Ppow + 16 * 4096, Ppow + 17 * 4096); // P17..P32
    mat_mul_batch<<<31, 256, 0, stream>>>(Ppow + 4096, Ppow + 32 * 4096, Ppow + 33 * 4096); // P33..P63
    mat_mul_batch<<<1,  256, 0, stream>>>(Ppow + 32 * 4096, Ppow + 32 * 4096, A64);         // A^64

    fused_gemm   <<<1024, 512,  0, stream>>>(x, bvec, bfrag, vecs, vecs + 64, vecs + 128, u);
    chunk_gemm   <<<512,  256,  0, stream>>>(u, Ppow, ps);
    final_combine<<<1,    1024, 0, stream>>>(ps, A64, C, out);
}
